// Head_32538672234833
// MI455X (gfx1250) — matrix operand order repaired and checked by reading, not compiled, not run on hardware
//
#include <hip/hip_runtime.h>
#include <hip/hip_bf16.h>
#include <stdint.h>

typedef __attribute__((ext_vector_type(16))) _Float16 v16h;
typedef __attribute__((ext_vector_type(8)))  float    v8f;
typedef __attribute__((ext_vector_type(4)))  int      i4vec;

#define EMB   768
#define HEAD  64
#define BATCH 4
#define SEQ   4096
#define ROWS  (BATCH*SEQ)

#define GLOBAL_AS __attribute__((address_space(1)))
#define LDS_AS    __attribute__((address_space(3)))

#if defined(__has_builtin)
#  if __has_builtin(__builtin_amdgcn_global_load_async_to_lds_b128)
#    define HAVE_ASYNC_LDS 1
#  endif
#endif
#ifndef HAVE_ASYNC_LDS
#  define HAVE_ASYNC_LDS 0
#endif

// 16-bit A-matrix (16x32) layout, contiguous K order matching B: element pair j (j=0..7) of a lane
// covers K = 16*half + 2*j, +1.
__device__ __forceinline__ int a_pair_k(int j, int half) {
    return 16*half + 2*j;
}

__device__ __forceinline__ v8f wmma16(v16h a, v16h b, v8f c) {
    return __builtin_amdgcn_wmma_f32_16x16x32_f16(false, a, false, b, (short)0, c, false, false);
}

__device__ __forceinline__ void async_wait_all() {
#if HAVE_ASYNC_LDS
#  if __has_builtin(__builtin_amdgcn_s_wait_asynccnt)
    __builtin_amdgcn_s_wait_asynccnt(0);
#  else
    asm volatile("s_wait_asynccnt 0" ::: "memory");
#  endif
#endif
}

// ---------------- Kernel 0: weights f32 -> f16 ----------------
__global__ void wcvt_kernel(const float* __restrict__ w, _Float16* __restrict__ o, int n) {
    int i = blockIdx.x * blockDim.x + threadIdx.x;
    if (i < n) o[i] = (_Float16)w[i];
}

// ---------------- Kernel 1: QKV projection ----------------
// grid = ROWS/16, block = 96 (3 waves). Wave0 -> Q (scaled), wave1 -> K, wave2 -> V (transposed).
__global__ __launch_bounds__(96) void qkv_proj_kernel(
    const float* __restrict__ ctx,
    const _Float16* __restrict__ wq, const _Float16* __restrict__ wk, const _Float16* __restrict__ wv,
    _Float16* __restrict__ Q, _Float16* __restrict__ Kmat, _Float16* __restrict__ Vt)
{
    constexpr int STRIDE = EMB + 16;              // pad to dodge LDS bank conflicts
    __shared__ _Float16 As[16 * STRIDE];

    const int tid   = threadIdx.x;
    const int mbase = blockIdx.x * 16;

    // cooperative stage: 16x768 f32 tile -> f16 LDS (float4 vectorized)
    for (int i = tid; i < 16 * (EMB / 4); i += 96) {
        int row = i / (EMB / 4);
        int c4  = i % (EMB / 4);
        const float4 f = *(const float4*)(ctx + (size_t)(mbase + row) * EMB + c4 * 4);
        _Float16* d = &As[row * STRIDE + c4 * 4];
        d[0] = (_Float16)f.x; d[1] = (_Float16)f.y; d[2] = (_Float16)f.z; d[3] = (_Float16)f.w;
    }
    __syncthreads();

    const int w    = tid >> 5;
    const int lane = tid & 31;
    const int n    = lane & 15;
    const int half = lane >> 4;

    const _Float16* W = (w == 0) ? wq : (w == 1) ? wk : wv;

    v8f c0 = {}, c1 = {}, c2 = {}, c3 = {};
    for (int kk = 0; kk < EMB; kk += 32) {
        union { v16h v; uint32_t u[8]; } a;
        #pragma unroll
        for (int j = 0; j < 8; ++j)
            a.u[j] = *(const uint32_t*)&As[n * STRIDE + kk + a_pair_k(j, half)];

        // B = W^T tile: lane column = head dim, contiguous 32B per lane from W rows
        v16h b0 = *(const v16h*)(W + (size_t)(n      ) * EMB + kk + 16 * half);
        v16h b1 = *(const v16h*)(W + (size_t)(n + 16 ) * EMB + kk + 16 * half);
        v16h b2 = *(const v16h*)(W + (size_t)(n + 32 ) * EMB + kk + 16 * half);
        v16h b3 = *(const v16h*)(W + (size_t)(n + 48 ) * EMB + kk + 16 * half);

        c0 = wmma16(a.v, b0, c0);
        c1 = wmma16(a.v, b1, c1);
        c2 = wmma16(a.v, b2, c2);
        c3 = wmma16(a.v, b3, c3);
    }

    if (w < 2) {
        const float scale = (w == 0) ? 0.03608439182435161f : 1.0f; // 1/sqrt(768) folded into Q
        _Float16* out = (w == 0) ? Q : Kmat;
        #pragma unroll
        for (int r = 0; r < 8; ++r) {
            size_t row = (size_t)(mbase + r + 8 * half) * HEAD;
            out[row + n     ] = (_Float16)(c0[r] * scale);
            out[row + n + 16] = (_Float16)(c1[r] * scale);
            out[row + n + 32] = (_Float16)(c2[r] * scale);
            out[row + n + 48] = (_Float16)(c3[r] * scale);
        }
    } else {
        // V transposed: Vt[b][h][t] -> per lane 8 contiguous t's per head row
        const int b    = mbase / SEQ;
        const int tloc = mbase % SEQ;
        #pragma unroll
        for (int r = 0; r < 8; ++r) {
            int t = tloc + r + 8 * half;
            Vt[((size_t)b * HEAD + n     ) * SEQ + t] = (_Float16)c0[r];
            Vt[((size_t)b * HEAD + n + 16) * SEQ + t] = (_Float16)c1[r];
            Vt[((size_t)b * HEAD + n + 32) * SEQ + t] = (_Float16)c2[r];
            Vt[((size_t)b * HEAD + n + 48) * SEQ + t] = (_Float16)c3[r];
        }
    }
}

// ---------------- Kernel 2: causal flash attention ----------------
// grid = BATCH*(SEQ/64), block = 128 (4 waves); wave w owns 16 query rows.
// K/V tiles (64 keys) staged once per block in LDS (async copy + double buffer),
// shared by all 4 waves -> 4x less L2 traffic than per-wave loads.

#define KV_PAD    8
#define KV_STRIDE (HEAD + KV_PAD)   // 72 halfs = 144B row stride: conflict-free b128 reads

__device__ __forceinline__ void stage_kv(const _Float16* __restrict__ Kp,
                                         const _Float16* __restrict__ Vp,
                                         int s0, int tid,
                                         _Float16* ksBuf, _Float16* vsBuf)
{
    // 2 matrices x 64 rows x 64 halfs; 128 threads x 4 iters x 16B per copy each
    #pragma unroll
    for (int it = 0; it < 4; ++it) {
        int idx = tid + it * 128;        // 0..511
        int row = idx >> 3;              // 0..63  (key index for K, head index for Vt)
        int col = (idx & 7) << 3;        // 0,8,...,56 halfs
        const _Float16* gk = Kp + (size_t)(s0 + row) * HEAD + col;
        const _Float16* gv = Vp + (size_t)row * SEQ + s0 + col;
        _Float16* lk = ksBuf + row * KV_STRIDE + col;
        _Float16* lv = vsBuf + row * KV_STRIDE + col;
#if HAVE_ASYNC_LDS
        __builtin_amdgcn_global_load_async_to_lds_b128(
            (GLOBAL_AS i4vec*)gk, (LDS_AS i4vec*)lk, 0, 0);
        __builtin_amdgcn_global_load_async_to_lds_b128(
            (GLOBAL_AS i4vec*)gv, (LDS_AS i4vec*)lv, 0, 0);
#else
        *(uint4*)lk = *(const uint4*)gk;
        *(uint4*)lv = *(const uint4*)gv;
#endif
    }
}

__global__ __launch_bounds__(128) void attn_kernel(
    const _Float16* __restrict__ Q, const _Float16* __restrict__ Kmat,
    const _Float16* __restrict__ Vt, float* __restrict__ out)
{
    __shared__ _Float16 Ks[2][64][KV_STRIDE];     // [buf][key][head]
    __shared__ _Float16 Vs[2][HEAD][KV_STRIDE];   // [buf][head][key]
    __shared__ _Float16 Pst[4][16][72];           // per-wave P relayout (D->A), padded

    const int tid  = threadIdx.x;
    const int w    = tid >> 5;
    const int lane = tid & 31;
    const int n    = lane & 15;
    const int half = lane >> 4;

    const int b      = blockIdx.x >> 6;     // / (SEQ/64)
    const int qt     = blockIdx.x & 63;
    const int qblock = qt * 64;             // block's first query row (in batch)
    const int qbase  = qblock + w * 16;     // this wave's first query row

    const _Float16* Qp = Q    + ((size_t)b * SEQ + qbase) * HEAD;
    const _Float16* Kp = Kmat + (size_t)b * SEQ * HEAD;
    const _Float16* Vp = Vt   + (size_t)b * HEAD * SEQ;

    // Q A-fragments (K = 0..31 and 32..63 of the head dim), loaded once
    union { v16h v; uint32_t u[8]; } aq0, aq1;
    #pragma unroll
    for (int j = 0; j < 8; ++j) {
        aq0.u[j] = *(const uint32_t*)(Qp + (size_t)n * HEAD +      a_pair_k(j, half));
        aq1.u[j] = *(const uint32_t*)(Qp + (size_t)n * HEAD + 32 + a_pair_k(j, half));
    }

    v8f o0 = {}, o1 = {}, o2 = {}, o3 = {};
    float mrow[8], lsum[8];
    #pragma unroll
    for (int r = 0; r < 8; ++r) { mrow[r] = -1e30f; lsum[r] = 0.0f; }

    const int keyEnd = qblock + 64;         // keys needed by the whole block
    int cur = 0;
    stage_kv(Kp, Vp, 0, tid, &Ks[0][0][0], &Vs[0][0][0]);

    for (int s0 = 0; s0 < keyEnd; s0 += 64, cur ^= 1) {
        async_wait_all();
        __syncthreads();                    // buffer `cur` ready; prev buffer free
        if (s0 + 64 < keyEnd)
            stage_kv(Kp, Vp, s0 + 64, tid, &Ks[cur ^ 1][0][0], &Vs[cur ^ 1][0][0]);

        if (s0 < qbase + 16) {              // wave-uniform: this wave needs this tile
            // S = Q K^T for 4 chunks of 16 keys (K-dim = 64)
            v8f st[4];
            #pragma unroll
            for (int nc = 0; nc < 4; ++nc) {
                v16h bk0 = *(const v16h*)&Ks[cur][nc * 16 + n][     16 * half];
                v16h bk1 = *(const v16h*)&Ks[cur][nc * 16 + n][32 + 16 * half];
                v8f s = {};
                s = wmma16(aq0.v, bk0, s);
                s = wmma16(aq1.v, bk1, s);
                st[nc] = s;
            }

            // causal mask + online softmax over 64 keys (reductions in 16-lane groups)
            float p[4][8];
            #pragma unroll
            for (int r = 0; r < 8; ++r) {
                int t = qbase + r + 8 * half;
                #pragma unroll
                for (int nc = 0; nc < 4; ++nc) {
                    float v = st[nc][r];
                    if (s0 + nc * 16 + n > t) v = -1e30f;
                    p[nc][r] = v;
                }
                float mx = fmaxf(fmaxf(p[0][r], p[1][r]), fmaxf(p[2][r], p[3][r]));
                mx = fmaxf(mx, __shfl_xor(mx, 1, 32));
                mx = fmaxf(mx, __shfl_xor(mx, 2, 32));
                mx = fmaxf(mx, __shfl_xor(mx, 4, 32));
                mx = fmaxf(mx, __shfl_xor(mx, 8, 32));

                float newm = fmaxf(mrow[r], mx);
                float sc   = __expf(mrow[r] - newm);
                float sum  = 0.0f;
                #pragma unroll
                for (int nc = 0; nc < 4; ++nc) {
                    p[nc][r] = __expf(p[nc][r] - newm);
                    sum += p[nc][r];
                }
                sum += __shfl_xor(sum, 1, 32);
                sum += __shfl_xor(sum, 2, 32);
                sum += __shfl_xor(sum, 4, 32);
                sum += __shfl_xor(sum, 8, 32);

                lsum[r] = lsum[r] * sc + sum;
                mrow[r] = newm;
                o0[r] *= sc; o1[r] *= sc; o2[r] *= sc; o3[r] *= sc;
            }

            // relayout P: D-layout -> A-layout via per-wave LDS (in-wave, no barrier)
            #pragma unroll
            for (int r = 0; r < 8; ++r) {
                #pragma unroll
                for (int nc = 0; nc < 4; ++nc)
                    Pst[w][r + 8 * half][nc * 16 + n] = (_Float16)p[nc][r];
            }
            union { v16h v; uint32_t u[8]; } pa0, pa1;
            #pragma unroll
            for (int j = 0; j < 8; ++j) {
                pa0.u[j] = *(const uint32_t*)&Pst[w][n][     a_pair_k(j, half)];
                pa1.u[j] = *(const uint32_t*)&Pst[w][n][32 + a_pair_k(j, half)];
            }

            // O += P V  (keys 0..31 via pa0, 32..63 via pa1; Vs is [head][key])
            v16h bv;
            bv = *(const v16h*)&Vs[cur][n     ][     16 * half]; o0 = wmma16(pa0.v, bv, o0);
            bv = *(const v16h*)&Vs[cur][n     ][32 + 16 * half]; o0 = wmma16(pa1.v, bv, o0);
            bv = *(const v16h*)&Vs[cur][n + 16][     16 * half]; o1 = wmma16(pa0.v, bv, o1);
            bv = *(const v16h*)&Vs[cur][n + 16][32 + 16 * half]; o1 = wmma16(pa1.v, bv, o1);
            bv = *(const v16h*)&Vs[cur][n + 32][     16 * half]; o2 = wmma16(pa0.v, bv, o2);
            bv = *(const v16h*)&Vs[cur][n + 32][32 + 16 * half]; o2 = wmma16(pa1.v, bv, o2);
            bv = *(const v16h*)&Vs[cur][n + 48][     16 * half]; o3 = wmma16(pa0.v, bv, o3);
            bv = *(const v16h*)&Vs[cur][n + 48][32 + 16 * half]; o3 = wmma16(pa1.v, bv, o3);
        }
    }

    #pragma unroll
    for (int r = 0; r < 8; ++r) {
        float inv = 1.0f / lsum[r];
        size_t row = ((size_t)b * SEQ + qbase + r + 8 * half) * HEAD;
        out[row + n     ] = o0[r] * inv;
        out[row + n + 16] = o1[r] * inv;
        out[row + n + 32] = o2[r] * inv;
        out[row + n + 48] = o3[r] * inv;
    }
}

extern "C" void kernel_launch(void* const* d_in, const int* in_sizes, int n_in,
                              void* d_out, int out_size, void* d_ws, size_t ws_size,
                              hipStream_t stream) {
    const float* ctx = (const float*)d_in[0];
    const float* Wk  = (const float*)d_in[1];  // setup_inputs order: context, Wk, Wq, Wv
    const float* Wq  = (const float*)d_in[2];
    const float* Wv  = (const float*)d_in[3];
    float* out = (float*)d_out;

    _Float16* ws = (_Float16*)d_ws;
    const size_t qkvN = (size_t)ROWS * HEAD;       // 1,048,576 halfs each
    _Float16* Q16  = ws;
    _Float16* K16  = ws + qkvN;
    _Float16* Vt16 = ws + 2 * qkvN;
    _Float16* wq16 = ws + 3 * qkvN;
    _Float16* wk16 = wq16 + HEAD * EMB;
    _Float16* wv16 = wk16 + HEAD * EMB;

    const int wn = HEAD * EMB;
    wcvt_kernel<<<(wn + 255) / 256, 256, 0, stream>>>(Wq, wq16, wn);
    wcvt_kernel<<<(wn + 255) / 256, 256, 0, stream>>>(Wk, wk16, wn);
    wcvt_kernel<<<(wn + 255) / 256, 256, 0, stream>>>(Wv, wv16, wn);

    qkv_proj_kernel<<<ROWS / 16, 96, 0, stream>>>(ctx, wq16, wk16, wv16, Q16, K16, Vt16);

    attn_kernel<<<BATCH * (SEQ / 64), 128, 0, stream>>>(Q16, K16, Vt16, out);
}
